// ptBEVnet_54580444398347
// MI455X (gfx1250) — matrix operand order repaired and checked by reading, not compiled, not run on hardware
//
#include <hip/hip_runtime.h>

// ---------------------------------------------------------------------------
// PolarSeg ptBEVnet for MI455X (gfx1250, wave32, WMMA):
//   BN0 -> [Lin+BN+ReLU]x3 -> Lin -> scatter_max -> maxpool3x3
// Strategy:
//   * bf16 activations/weights, f32 WMMA accumulation (v_wmma_f32_16x16x32_bf16)
//   * BN folded into per-channel scale/shift applied while staging tiles in LDS
//   * A/B fragments assembled as dword vectors (bit-cast to v16bf); A-fragments
//     hoisted out of the N-tile loop (register-cached across all K steps)
//   * uniform full-tile fast path in the epilogue (no per-lane exec churn)
//   * per-layer kernels (BN stats force global sync); stats via LDS ds_add_f32
//     reduction + one global atomic per channel per block
//   * layer4 epilogue scatters directly with monotonic-encoded u32 atomicMax
//   * maxpool runs in the encoded domain (order-preserving map), decodes at end
// ---------------------------------------------------------------------------

typedef __attribute__((ext_vector_type(16))) __bf16        v16bf;
typedef __attribute__((ext_vector_type(8)))  float         v8f;
typedef __attribute__((ext_vector_type(8)))  unsigned int  v8u;

constexpr int   BQ = 4, GH = 480, GW = 360;
constexpr int   COUT_FINAL = 64;
constexpr float EPS = 1e-5f;

// ---- workspace layout (bytes); h-buffer offsets computed at launch from N ----
constexpr size_t OFF_STATS = 0;        // 1024 f32: packed per-layer sum/sumsq
constexpr size_t OFF_WT1   = 4096;     // 64  x 32  bf16 (K padded to 32, zeros)
constexpr size_t OFF_WT2   = 8192;     // 128 x 64  bf16
constexpr size_t OFF_WT3   = 24576;    // 256 x 128 bf16
constexpr size_t OFF_WT4   = 90112;    // 64  x 256 bf16
constexpr size_t OFF_H1    = 122880;   // then H1(N*64*2) H2(N*128*2) H3(N*256*2) GRID

// monotonic float<->u32 map so unsigned max == float max
static __device__ __forceinline__ unsigned enc_f32(float f) {
    unsigned u = __float_as_uint(f);
    return (u & 0x80000000u) ? ~u : (u | 0x80000000u);
}
static __device__ __forceinline__ float dec_f32(unsigned u) {
    return (u & 0x80000000u) ? __uint_as_float(u ^ 0x80000000u)
                             : __uint_as_float(~u);
}

// ---------------------------------------------------------------------------
__global__ void zero_u32(unsigned* __restrict__ p, size_t n) {
    size_t i = (size_t)blockIdx.x * blockDim.x + threadIdx.x;
    if (i < n) p[i] = 0u;
}
__global__ void zero_f32(float* __restrict__ p, int n) {
    int i = blockIdx.x * blockDim.x + threadIdx.x;
    if (i < n) p[i] = 0.f;
}

// Transpose + f32->bf16 convert weights: W[i,o] (CIN x COUT) -> wbT[o, i_pad]
template <int CIN, int COUT>
__global__ void convert_w(const float* __restrict__ W, __bf16* __restrict__ wbT) {
    constexpr int CIN_P = ((CIN + 31) / 32) * 32;
    int e = blockIdx.x * blockDim.x + threadIdx.x;
    if (e >= COUT * CIN_P) return;
    int i = e % CIN_P, o = e / CIN_P;
    float v = (i < CIN) ? W[(size_t)i * COUT + o] : 0.f;
    wbT[e] = (__bf16)v;
}

// Per-channel sum / sum-of-squares of the 9-channel input features.
__global__ __launch_bounds__(256) void stats_input(const float* __restrict__ x,
                                                   float* __restrict__ s,
                                                   float* __restrict__ q, int n) {
    __shared__ float ss[9], sq[9];
    int tid = threadIdx.x;
    if (tid < 9) { ss[tid] = 0.f; sq[tid] = 0.f; }
    __syncthreads();
    float ls[9], lq[9];
#pragma unroll
    for (int c = 0; c < 9; ++c) { ls[c] = 0.f; lq[c] = 0.f; }
    for (size_t r = (size_t)blockIdx.x * 256 + tid; r < (size_t)n;
         r += (size_t)gridDim.x * 256) {
#pragma unroll
        for (int c = 0; c < 9; ++c) {
            float v = x[r * 9 + c];
            ls[c] += v; lq[c] += v * v;
        }
    }
#pragma unroll
    for (int c = 0; c < 9; ++c) { atomicAdd(&ss[c], ls[c]); atomicAdd(&sq[c], lq[c]); }
    __syncthreads();
    if (tid < 9) { atomicAdd(&s[tid], ss[tid]); atomicAdd(&q[tid], sq[tid]); }
}

// ---------------------------------------------------------------------------
// Fused layer: normalize(+ReLU) input tile -> LDS (bf16) -> WMMA GEMM ->
//   MODE 0: store raw pre-BN output + accumulate next-layer BN stats
//   MODE 1: encode + global_atomic_max_u32 scatter into BEV grid
// 256 threads = 8 wave32 waves. BLOCK_M rows per block, 16x16 tiles per wave.
// ---------------------------------------------------------------------------
template <int CIN, int COUT, bool RELU_IN, bool IN_F32, int MODE, int BLOCK_M>
__global__ __launch_bounds__(256) void mlp_layer(
    const float*  __restrict__ in_f32, const __bf16* __restrict__ in_bf,
    const float*  __restrict__ in_sum, const float*  __restrict__ in_sq,
    const float*  __restrict__ gamma,  const float*  __restrict__ beta,
    const __bf16* __restrict__ wbT,    const float*  __restrict__ bias,
    __bf16* __restrict__ h_out, float* __restrict__ out_sum, float* __restrict__ out_sq,
    const int* __restrict__ pt_ind, unsigned* __restrict__ grid_enc, int nrows) {

    constexpr int CIN_P   = ((CIN + 31) / 32) * 32;  // K padded to WMMA step
    constexpr int KPAD    = CIN_P + 2;               // +1 dword: no LDS bank conflicts
    constexpr int M_TILES = BLOCK_M / 16;
    constexpr int NSPLIT  = 8 / M_TILES;             // waves sharing an M tile
    constexpr int NT      = COUT / 16;
    constexpr int KSTEPS  = CIN_P / 32;

    __shared__ __bf16 s_act[BLOCK_M * KPAD];
    __shared__ float  s_scale[CIN], s_shift[CIN];
    __shared__ float  s_sum[COUT], s_sq[COUT];

    const int   tid  = threadIdx.x;
    const int   row0 = blockIdx.x * BLOCK_M;
    const float invN = 1.0f / (float)nrows;

    // Fold BN of the *input* into per-channel scale/shift.
    if (tid < CIN) {
        float m  = in_sum[tid] * invN;
        float v  = in_sq[tid] * invN - m * m;
        float rs = rsqrtf(v + EPS);
        float sc = gamma[tid] * rs;
        s_scale[tid] = sc;
        s_shift[tid] = beta[tid] - m * sc;
    }
    if (MODE == 0 && tid < COUT) { s_sum[tid] = 0.f; s_sq[tid] = 0.f; }
    if (CIN % 32 != 0) {  // zero-pad K (layer1: CIN=9 -> 32)
        for (int e = tid; e < BLOCK_M * KPAD; e += 256) s_act[e] = (__bf16)0.f;
    }
    __syncthreads();

    // Stage normalized (+ReLU) activations into LDS as bf16.
    for (int e = tid; e < BLOCK_M * CIN; e += 256) {
        int row = e / CIN, c = e - row * CIN;
        int grow = row0 + row;
        float y = 0.f;
        if (grow < nrows) {
            float x;
            if constexpr (IN_F32) x = in_f32[(size_t)grow * CIN + c];
            else                  x = (float)in_bf[(size_t)grow * CIN + c];
            y = x * s_scale[c] + s_shift[c];
            if (RELU_IN) y = fmaxf(y, 0.f);
        }
        s_act[row * KPAD + c] = (__bf16)y;
    }
    __syncthreads();

    const int wave = tid >> 5, lane = tid & 31;
    const int half = lane >> 4, mloc = lane & 15;    // mloc doubles as B's n index
    const int mTile = wave % M_TILES;
    const int ntStart = wave / M_TILES;
    const bool fullTile = (row0 + BLOCK_M) <= nrows; // uniform across block

    // Preload ALL A fragments (reused by every N tile). Dword-pair aligned:
    // fragment dword v maps to K-pair at dword index v + 4*half (+4 for v>=4).
    v16bf afr[KSTEPS];
#pragma unroll
    for (int ks = 0; ks < KSTEPS; ++ks) {
        const unsigned* ad = (const unsigned*)&s_act[(mTile * 16 + mloc) * KPAD + ks * 32];
        v8u au;
#pragma unroll
        for (int v = 0; v < 8; ++v)
            au[v] = ad[v + 4 * half + ((v >= 4) ? 4 : 0)];
        afr[ks] = __builtin_bit_cast(v16bf, au);
    }

    for (int nt = ntStart; nt < NT; nt += NSPLIT) {
        v8f acc;
#pragma unroll
        for (int i = 0; i < 8; ++i) acc[i] = 0.f;

#pragma unroll
        for (int ks = 0; ks < KSTEPS; ++ks) {
            // B fragment (32x16 bf16): lane holds column n=mloc; lanes 0-15
            // cover K=0..7,16..23, lanes 16-31 K=8..15,24..31 (matches A).
            const unsigned* bd = (const unsigned*)
                &wbT[(size_t)(nt * 16 + mloc) * CIN_P + ks * 32 + 8 * half];
            v8u bu;
#pragma unroll
            for (int r = 0; r < 8; ++r) bu[r] = bd[r + ((r >= 4) ? 4 : 0)];
            v16bf b = __builtin_bit_cast(v16bf, bu);
            acc = __builtin_amdgcn_wmma_f32_16x16x32_bf16(
                false, afr[ks], false, b, (short)0, acc, false, false);
        }

        // C/D layout: VGPR r -> row r + 8*half, col = lane&15.
        const int   c  = nt * 16 + mloc;
        const float bz = bias[c];
        if constexpr (MODE == 0) {
            float lsum = 0.f, lsq = 0.f;
            if (fullTile) {
#pragma unroll
                for (int r = 0; r < 8; ++r) {
                    int grow = row0 + mTile * 16 + r + 8 * half;
                    float hv = acc[r] + bz;
                    h_out[(size_t)grow * COUT + c] = (__bf16)hv;
                    lsum += hv; lsq += hv * hv;
                }
            } else {
#pragma unroll
                for (int r = 0; r < 8; ++r) {
                    int grow = row0 + mTile * 16 + r + 8 * half;
                    if (grow < nrows) {
                        float hv = acc[r] + bz;
                        h_out[(size_t)grow * COUT + c] = (__bf16)hv;
                        lsum += hv; lsq += hv * hv;
                    }
                }
            }
            atomicAdd(&s_sum[c], lsum);   // ds_add_f32
            atomicAdd(&s_sq[c], lsq);
        } else {
#pragma unroll
            for (int r = 0; r < 8; ++r) {
                int grow = row0 + mTile * 16 + r + 8 * half;
                if (fullTile || grow < nrows) {
                    float hv = acc[r] + bz;
                    int bi = pt_ind[3 * (size_t)grow + 0];
                    int xi = pt_ind[3 * (size_t)grow + 1];
                    int yi = pt_ind[3 * (size_t)grow + 2];
                    size_t seg = ((size_t)bi * GH + xi) * GW + yi;
                    atomicMax(&grid_enc[seg * COUT_FINAL + c], enc_f32(hv));
                }
            }
        }
    }

    if constexpr (MODE == 0) {
        __syncthreads();
        if (tid < COUT) {
            atomicAdd(&out_sum[tid], s_sum[tid]);
            atomicAdd(&out_sq[tid], s_sq[tid]);
        }
    }
}

// ---------------------------------------------------------------------------
// Decode + 3x3/s1/p1 max-pool in the encoded (order-preserving) domain.
// Grid is [b,h,w,c] (c minor); output is [b,c,h,w] f32. Empty cells (u==0)
// map to enc(0.0f) = 0x80000000, matching the zero-filled reference grid.
// ---------------------------------------------------------------------------
__global__ __launch_bounds__(256) void maxpool_decode(const unsigned* __restrict__ g,
                                                      float* __restrict__ out) {
    size_t total = (size_t)BQ * COUT_FINAL * GH * GW;
    size_t idx = (size_t)blockIdx.x * 256 + threadIdx.x;
    if (idx >= total) return;
    int w = idx % GW;  size_t t = idx / GW;
    int h = t % GH;    t /= GH;
    int c = t % COUT_FINAL;
    int b = (int)(t / COUT_FINAL);
    unsigned best = 0u;  // below enc(-inf): identity for the included window
    for (int dh = -1; dh <= 1; ++dh) {
        int hh = h + dh; if (hh < 0 || hh >= GH) continue;
        for (int dw = -1; dw <= 1; ++dw) {
            int ww = w + dw; if (ww < 0 || ww >= GW) continue;
            unsigned v = g[(((size_t)b * GH + hh) * GW + ww) * COUT_FINAL + c];
            if (v == 0u) v = 0x80000000u;  // empty voxel -> 0.0f
            if (v > best) best = v;
        }
    }
    out[idx] = dec_f32(best);
}

// ---------------------------------------------------------------------------
extern "C" void kernel_launch(void* const* d_in, const int* in_sizes, int n_in,
                              void* d_out, int out_size, void* d_ws, size_t ws_size,
                              hipStream_t stream) {
    const float* pt_fea = (const float*)d_in[0];
    const int*   pt_ind = (const int*)d_in[1];
    const float* g0  = (const float*)d_in[2];
    const float* b0  = (const float*)d_in[3];
    const float* W1  = (const float*)d_in[4];
    const float* bl1 = (const float*)d_in[5];
    const float* g1  = (const float*)d_in[6];
    const float* be1 = (const float*)d_in[7];
    const float* W2  = (const float*)d_in[8];
    const float* bl2 = (const float*)d_in[9];
    const float* g2  = (const float*)d_in[10];
    const float* be2 = (const float*)d_in[11];
    const float* W3  = (const float*)d_in[12];
    const float* bl3 = (const float*)d_in[13];
    const float* g3  = (const float*)d_in[14];
    const float* be3 = (const float*)d_in[15];
    const float* W4  = (const float*)d_in[16];
    const float* bl4 = (const float*)d_in[17];

    const int N = in_sizes[0] / 9;

    char*  ws    = (char*)d_ws;
    float* stats = (float*)(ws + OFF_STATS);
    float *S0 = stats + 0,   *Q0 = stats + 16;
    float *S1 = stats + 32,  *Q1 = stats + 96;
    float *S2 = stats + 160, *Q2 = stats + 288;
    float *S3 = stats + 416, *Q3 = stats + 672;
    __bf16* WT1 = (__bf16*)(ws + OFF_WT1);
    __bf16* WT2 = (__bf16*)(ws + OFF_WT2);
    __bf16* WT3 = (__bf16*)(ws + OFF_WT3);
    __bf16* WT4 = (__bf16*)(ws + OFF_WT4);
    size_t offH1 = OFF_H1;
    size_t offH2 = offH1 + (size_t)N * 64 * 2;
    size_t offH3 = offH2 + (size_t)N * 128 * 2;
    size_t offG  = offH3 + (size_t)N * 256 * 2;
    __bf16*   H1   = (__bf16*)(ws + offH1);
    __bf16*   H2   = (__bf16*)(ws + offH2);
    __bf16*   H3   = (__bf16*)(ws + offH3);
    unsigned* GRID = (unsigned*)(ws + offG);

    const size_t gridElems = (size_t)BQ * GH * GW * COUT_FINAL;

    zero_f32<<<4, 256, 0, stream>>>(stats, 1024);
    zero_u32<<<(unsigned)((gridElems + 255) / 256), 256, 0, stream>>>(GRID, gridElems);
    convert_w<9, 64>   <<<(64 * 32 + 255) / 256,   256, 0, stream>>>(W1, WT1);
    convert_w<64, 128> <<<(128 * 64 + 255) / 256,  256, 0, stream>>>(W2, WT2);
    convert_w<128, 256><<<(256 * 128 + 255) / 256, 256, 0, stream>>>(W3, WT3);
    convert_w<256, 64> <<<(64 * 256 + 255) / 256,  256, 0, stream>>>(W4, WT4);
    stats_input<<<1024, 256, 0, stream>>>(pt_fea, S0, Q0, N);

    const int blocks128 = (N + 127) / 128;
    const int blocks64  = (N + 63) / 64;
    mlp_layer<9, 64, false, true, 0, 128><<<blocks128, 256, 0, stream>>>(
        pt_fea, nullptr, S0, Q0, g0, b0, WT1, bl1, H1, S1, Q1, nullptr, nullptr, N);
    mlp_layer<64, 128, true, false, 0, 128><<<blocks128, 256, 0, stream>>>(
        nullptr, H1, S1, Q1, g1, be1, WT2, bl2, H2, S2, Q2, nullptr, nullptr, N);
    mlp_layer<128, 256, true, false, 0, 128><<<blocks128, 256, 0, stream>>>(
        nullptr, H2, S2, Q2, g2, be2, WT3, bl3, H3, S3, Q3, nullptr, nullptr, N);
    mlp_layer<256, 64, true, false, 1, 64><<<blocks64, 256, 0, stream>>>(
        nullptr, H3, S3, Q3, g3, be3, WT4, bl4, nullptr, nullptr, nullptr,
        pt_ind, GRID, N);

    const size_t total = (size_t)BQ * COUT_FINAL * GH * GW;
    maxpool_decode<<<(unsigned)((total + 255) / 256), 256, 0, stream>>>(
        GRID, (float*)d_out);
}
